// MultiScalePointNet_51316269253474
// MI455X (gfx1250) — compile-verified, matrix operand order checked by reading, not run on hardware
//
#include <hip/hip_runtime.h>

typedef _Float16 v8h  __attribute__((ext_vector_type(8)));
typedef _Float16 v16h __attribute__((ext_vector_type(16)));
typedef float    v8f  __attribute__((ext_vector_type(8)));

#define NPTS 100000
#define KNB  20

__device__ __forceinline__ float leaky(float x) { return fmaxf(x, 0.2f * x); }

// Load a 16x32 f16 fragment (A-layout, or B == row-major W[out][in]) for wave32.
// p points at element [row = lane&15][kt*32]. Lane<16 takes K {0..7,16..23},
// lane>=16 takes K {8..15,24..31}, matching the CDNA5 16-bit A VGPR layout.
__device__ __forceinline__ v16h frag_ld(const _Float16* p, int lane) {
  const int koff = (lane & 16) ? 8 : 0;
  v8h lo = *(const v8h*)(p + koff);
  v8h hi = *(const v8h*)(p + koff + 16);
  return __builtin_shufflevector(lo, hi, 0,1,2,3,4,5,6,7,8,9,10,11,12,13,14,15);
}

__device__ __forceinline__ v8f wmma_f16(v16h a, v16h b, v8f c) {
  return __builtin_amdgcn_wmma_f32_16x16x32_f16(false, a, false, b, (short)0, c, false, false);
}

// ---------------- weight fp32 -> fp16 (with K padding) ----------------
__global__ void cvt_pad_kernel(const float* __restrict__ src, _Float16* __restrict__ dst,
                               int rows, int inK, int padK) {
  int i = blockIdx.x * blockDim.x + threadIdx.x;
  if (i >= rows * padK) return;
  int r = i / padK, c = i - r * padK;
  dst[i] = (c < inK) ? (_Float16)src[r * inK + c] : (_Float16)0.0f;
}

// ---------------- per-scale patch PointNet (3->64->128->256, max-pool) ----------------
// S (neighbor count) and SC (scale index) are compile-time so all the r/S, r%S
// in the gather and the max-pool scatter fold to mul-shift instead of idiv.
template <int S, int SC>
__global__ __launch_bounds__(128)
void patch_kernel(const float* __restrict__ points, const int* __restrict__ nbr,
                  const _Float16* __restrict__ w1h, const float* __restrict__ pb1,
                  const _Float16* __restrict__ w2h, const float* __restrict__ pb2,
                  const _Float16* __restrict__ w3h, const float* __restrict__ pb3,
                  _Float16* __restrict__ feats) {
  __shared__ __align__(16) _Float16 stage[4][4096]; // 8KB per wave
  __shared__ float pooled[16 * 256];                // 16 points x 256 feats

  const int tid  = threadIdx.x;
  const int lane = tid & 31;
  const int wv   = tid >> 5;
  const int P0   = blockIdx.x * 16;

  for (int i = tid; i < 16 * 256; i += 128) pooled[i] = -3.0e38f;
  __syncthreads();

  _Float16* A0 = &stage[wv][0];     // [16][32]
  _Float16* A1 = &stage[wv][512];   // [16][64]
  _Float16* A2 = &stage[wv][1536];  // [16][128]
  const int m   = lane & 15;
  const int hi8 = (lane & 16) ? 8 : 0;

  const _Float16* w1s = w1h + SC * 64 * 32;
  const _Float16* w2s = w2h + SC * 128 * 64;
  const _Float16* w3s = w3h + SC * 256 * 128;
  const float bias1 = pb1[SC * 64 + m];   // per-lane column bias, tiles add dt*16
  // (full bias gathered per tile below; only the lane column index is fixed)

  for (int t = wv; t < S; t += 4) {
    // ---- stage A0 = centered neighbor coords, zero-padded K=3->32 ----
    uint4 z = make_uint4(0u, 0u, 0u, 0u);
    ((uint4*)A0)[lane * 2]     = z;
    ((uint4*)A0)[lane * 2 + 1] = z;
    if (lane < 16) {
      int r  = t * 16 + lane;
      int pl = r / S;
      int j  = r - pl * S;
      int p  = P0 + pl;
      int idx = nbr[p * KNB + j];
      float cx = points[p * 3 + 0], cy = points[p * 3 + 1], cz = points[p * 3 + 2];
      A0[lane * 32 + 0] = (_Float16)(points[idx * 3 + 0] - cx);
      A0[lane * 32 + 1] = (_Float16)(points[idx * 3 + 1] - cy);
      A0[lane * 32 + 2] = (_Float16)(points[idx * 3 + 2] - cz);
    }

    // ---- layer 1: [16x32(pad)] x [32x64] ----
    v16h a0 = frag_ld(A0 + m * 32, lane);
#pragma unroll
    for (int dt = 0; dt < 4; ++dt) {
      v16h b = frag_ld(w1s + (dt * 16 + m) * 32, lane);
      v8f c = {};
      c = wmma_f16(a0, b, c);
      float bias = pb1[SC * 64 + dt * 16 + m];
#pragma unroll
      for (int i = 0; i < 8; ++i) {
        float v = leaky(c[i] + bias);
        A1[(i + hi8) * 64 + dt * 16 + m] = (_Float16)v;
      }
    }
    (void)bias1;

    // ---- layer 2: [16x64] x [64x128] ----
    v8f acc2[8] = {};
#pragma unroll
    for (int kt = 0; kt < 2; ++kt) {
      v16h a = frag_ld(A1 + m * 64 + kt * 32, lane);
#pragma unroll
      for (int dt = 0; dt < 8; ++dt)
        acc2[dt] = wmma_f16(a, frag_ld(w2s + (dt * 16 + m) * 64 + kt * 32, lane), acc2[dt]);
    }
#pragma unroll
    for (int dt = 0; dt < 8; ++dt) {
      float bias = pb2[SC * 128 + dt * 16 + m];
#pragma unroll
      for (int i = 0; i < 8; ++i) {
        float v = leaky(acc2[dt][i] + bias);
        A2[(i + hi8) * 128 + dt * 16 + m] = (_Float16)v;
      }
    }

    // ---- layer 3: [16x128] x [128x256], then max-pool over neighbor rows ----
    for (int half = 0; half < 2; ++half) {
      v8f acc[8] = {};
#pragma unroll
      for (int kt = 0; kt < 4; ++kt) {
        v16h a = frag_ld(A2 + m * 128 + kt * 32, lane);
#pragma unroll
        for (int dt = 0; dt < 8; ++dt) {
          int d = half * 8 + dt;
          acc[dt] = wmma_f16(a, frag_ld(w3s + (d * 16 + m) * 128 + kt * 32, lane), acc[dt]);
        }
      }
#pragma unroll
      for (int dt = 0; dt < 8; ++dt) {
        int col = (half * 8 + dt) * 16 + m;
#pragma unroll
        for (int i = 0; i < 8; ++i) {
          int r  = t * 16 + i + hi8;
          int pl = r / S;                                  // compile-time S -> mul-shift
          atomicMax(&pooled[pl * 256 + col], acc[dt][i]);  // ds float max
        }
      }
    }
  }
  __syncthreads();

  // bias3 commutes with the max; add it here and emit f16 features.
  for (int i = tid; i < 16 * 256; i += 128) {
    int pl = i >> 8, col = i & 255;
    float v = pooled[i] + pb3[SC * 256 + col];
    feats[(size_t)(P0 + pl) * 768 + SC * 256 + col] = (_Float16)v;
  }
}

// ---------------- shared MLP: 768->256->128->64->32->1 ----------------
__global__ __launch_bounds__(64)
void mlp_kernel(const _Float16* __restrict__ feats,
                const _Float16* __restrict__ s1h, const float* __restrict__ sb1,
                const _Float16* __restrict__ s2h, const float* __restrict__ sb2,
                const _Float16* __restrict__ s3h, const float* __restrict__ sb3,
                const _Float16* __restrict__ s4h, const float* __restrict__ sb4,
                const float* __restrict__ sw5, const float* __restrict__ sb5,
                float* __restrict__ out) {
  __shared__ __align__(16) _Float16 st[2][8192]; // 16KB per wave
  const int tid  = threadIdx.x;
  const int lane = tid & 31;
  const int wv   = tid >> 5;
  const int R0   = (blockIdx.x * 2 + wv) * 16;
  const int m    = lane & 15;
  const int hi8  = (lane & 16) ? 8 : 0;

  _Float16* H1 = &st[wv][0];     // [16][256]
  _Float16* H2 = &st[wv][4096];  // [16][128]
  _Float16* H3 = &st[wv][6144];  // [16][64]
  _Float16* H4 = &st[wv][7168];  // [16][32]

  // ---- layer 1: [16x768] x [768x256], A straight from global f16 feats ----
  const _Float16* arow = feats + (size_t)(R0 + m) * 768;
  for (int half = 0; half < 2; ++half) {
    v8f acc[8] = {};
    for (int kt = 0; kt < 24; ++kt) {
      v16h a = frag_ld(arow + kt * 32, lane);
#pragma unroll
      for (int dt = 0; dt < 8; ++dt) {
        int d = half * 8 + dt;
        acc[dt] = wmma_f16(a, frag_ld(s1h + (d * 16 + m) * 768 + kt * 32, lane), acc[dt]);
      }
    }
#pragma unroll
    for (int dt = 0; dt < 8; ++dt) {
      int d = half * 8 + dt;
      float bias = sb1[d * 16 + m];
#pragma unroll
      for (int i = 0; i < 8; ++i) {
        float v = leaky(acc[dt][i] + bias);
        H1[(i + hi8) * 256 + d * 16 + m] = (_Float16)v;
      }
    }
  }

  // ---- layer 2: 256 -> 128 ----
  {
    v8f acc[8] = {};
    for (int kt = 0; kt < 8; ++kt) {
      v16h a = frag_ld(H1 + m * 256 + kt * 32, lane);
#pragma unroll
      for (int dt = 0; dt < 8; ++dt)
        acc[dt] = wmma_f16(a, frag_ld(s2h + (dt * 16 + m) * 256 + kt * 32, lane), acc[dt]);
    }
#pragma unroll
    for (int dt = 0; dt < 8; ++dt) {
      float bias = sb2[dt * 16 + m];
#pragma unroll
      for (int i = 0; i < 8; ++i) {
        float v = leaky(acc[dt][i] + bias);
        H2[(i + hi8) * 128 + dt * 16 + m] = (_Float16)v;
      }
    }
  }

  // ---- layer 3: 128 -> 64 ----
  {
    v8f acc[4] = {};
#pragma unroll
    for (int kt = 0; kt < 4; ++kt) {
      v16h a = frag_ld(H2 + m * 128 + kt * 32, lane);
#pragma unroll
      for (int dt = 0; dt < 4; ++dt)
        acc[dt] = wmma_f16(a, frag_ld(s3h + (dt * 16 + m) * 128 + kt * 32, lane), acc[dt]);
    }
#pragma unroll
    for (int dt = 0; dt < 4; ++dt) {
      float bias = sb3[dt * 16 + m];
#pragma unroll
      for (int i = 0; i < 8; ++i) {
        float v = leaky(acc[dt][i] + bias);
        H3[(i + hi8) * 64 + dt * 16 + m] = (_Float16)v;
      }
    }
  }

  // ---- layer 4: 64 -> 32 ----
  {
    v8f acc[2] = {};
#pragma unroll
    for (int kt = 0; kt < 2; ++kt) {
      v16h a = frag_ld(H3 + m * 64 + kt * 32, lane);
#pragma unroll
      for (int dt = 0; dt < 2; ++dt)
        acc[dt] = wmma_f16(a, frag_ld(s4h + (dt * 16 + m) * 64 + kt * 32, lane), acc[dt]);
    }
#pragma unroll
    for (int dt = 0; dt < 2; ++dt) {
      float bias = sb4[dt * 16 + m];
#pragma unroll
      for (int i = 0; i < 8; ++i) {
        float v = leaky(acc[dt][i] + bias);
        H4[(i + hi8) * 32 + dt * 16 + m] = (_Float16)v;
      }
    }
  }

  // ---- layer 5: 32 -> 1 (VALU dot) ----
  if (lane < 16) {
    float sum = sb5[0];
#pragma unroll
    for (int c = 0; c < 32; ++c) sum += (float)H4[lane * 32 + c] * sw5[c];
    out[R0 + lane] = sum;
  }
}

// ---------------- host-side launch ----------------
extern "C" void kernel_launch(void* const* d_in, const int* in_sizes, int n_in,
                              void* d_out, int out_size, void* d_ws, size_t ws_size,
                              hipStream_t stream) {
  const float* points = (const float*)d_in[0];
  const int*   nbr    = (const int*)d_in[1];
  const float* pw1 = (const float*)d_in[2];
  const float* pb1 = (const float*)d_in[3];
  const float* pw2 = (const float*)d_in[4];
  const float* pb2 = (const float*)d_in[5];
  const float* pw3 = (const float*)d_in[6];
  const float* pb3 = (const float*)d_in[7];
  const float* sw1 = (const float*)d_in[8];
  const float* sb1 = (const float*)d_in[9];
  const float* sw2 = (const float*)d_in[10];
  const float* sb2 = (const float*)d_in[11];
  const float* sw3 = (const float*)d_in[12];
  const float* sb3 = (const float*)d_in[13];
  const float* sw4 = (const float*)d_in[14];
  const float* sb4 = (const float*)d_in[15];
  const float* sw5 = (const float*)d_in[16];
  const float* sb5 = (const float*)d_in[17];
  float* out = (float*)d_out;

  char* ws = (char*)d_ws;
  _Float16* w1h = (_Float16*)(ws + 0);       // 3*64*32
  _Float16* w2h = (_Float16*)(ws + 12288);   // 3*128*64
  _Float16* w3h = (_Float16*)(ws + 61440);   // 3*256*128
  _Float16* s1h = (_Float16*)(ws + 258048);  // 256*768
  _Float16* s2h = (_Float16*)(ws + 651264);  // 128*256
  _Float16* s3h = (_Float16*)(ws + 716800);  // 64*128
  _Float16* s4h = (_Float16*)(ws + 733184);  // 32*64
  _Float16* feats = (_Float16*)(ws + 737280); // N*768 f16

  auto cvt = [&](const float* src, _Float16* dst, int rows, int inK, int padK) {
    int n = rows * padK;
    cvt_pad_kernel<<<(n + 255) / 256, 256, 0, stream>>>(src, dst, rows, inK, padK);
  };
  cvt(pw1, w1h, 3 * 64, 3, 32);
  cvt(pw2, w2h, 3 * 128, 64, 64);
  cvt(pw3, w3h, 3 * 256, 128, 128);
  cvt(sw1, s1h, 256, 768, 768);
  cvt(sw2, s2h, 128, 256, 256);
  cvt(sw3, s3h, 64, 128, 128);
  cvt(sw4, s4h, 32, 64, 64);

  dim3 gridA(NPTS / 16);
  patch_kernel<5, 0><<<gridA, 128, 0, stream>>>(points, nbr, w1h, pb1, w2h, pb2, w3h, pb3, feats);
  patch_kernel<10, 1><<<gridA, 128, 0, stream>>>(points, nbr, w1h, pb1, w2h, pb2, w3h, pb3, feats);
  patch_kernel<20, 2><<<gridA, 128, 0, stream>>>(points, nbr, w1h, pb1, w2h, pb2, w3h, pb3, feats);

  mlp_kernel<<<NPTS / 32, 64, 0, stream>>>(feats, s1h, sb1, s2h, sb2, s3h, sb3,
                                           s4h, sb4, sw5, sb5, out);
}
